// AttentionHead_51316269252652
// MI455X (gfx1250) — compile-verified, matrix operand order checked by reading, not run on hardware
//
#include <hip/hip_runtime.h>
#include <hip/hip_bf16.h>

typedef __attribute__((ext_vector_type(16))) _Float16 v16h;
typedef __attribute__((ext_vector_type(8)))  _Float16 v8h;
typedef __attribute__((ext_vector_type(8)))  float    v8f;

#define DMODEL 1024
#define HEAD   64
#define SEQ    4096
#define NROWS  (4 * SEQ)
#define KCHUNK 256

__device__ __forceinline__ v16h cat16(v8h lo, v8h hi) {
  return __builtin_shufflevector(lo, hi, 0,1,2,3,4,5,6,7,8,9,10,11,12,13,14,15);
}

__device__ __forceinline__ v8f wmma_f16(v16h a, v16h b, v8f c) {
  return __builtin_amdgcn_wmma_f32_16x16x32_f16(
      /*neg_a=*/false, a, /*neg_b=*/false, b,
      /*c_mod=*/(short)0, c, /*reuse_a=*/false, /*reuse_b=*/false);
}

// ---------------------------------------------------------------------------
// Kernel 1: QKV projection.  y = x @ W  for W in {Wk, Wq, Wv} (blockIdx.y).
// 256 threads = 8 waves; each wave owns a 16-row tile (block covers 128 rows).
// W chunk (256 x 64) staged to LDS transposed ([n][k]); all four B fragments
// are loaded before the four WMMAs so DS latency overlaps matrix math.
// K,Q stored row-major f16; V stored transposed ([b][64][4096]).
// ---------------------------------------------------------------------------
__global__ __launch_bounds__(256) void qkv_proj_kernel(
    const float* __restrict__ x,
    const float* __restrict__ Wk, const float* __restrict__ Wq,
    const float* __restrict__ Wv,
    _Float16* __restrict__ Kf, _Float16* __restrict__ Qf,
    _Float16* __restrict__ Vt)
{
  __shared__ _Float16 lds_wt[HEAD * KCHUNK];   // 32 KB, transposed [n][k]

  const int which = blockIdx.y;
  const float* __restrict__ W = (which == 0) ? Wk : (which == 1 ? Wq : Wv);

  const int tid  = threadIdx.x;
  const int lane = tid & 31;
  const int wave = tid >> 5;
  const int m    = lane & 15;            // row (A) / col (B,C) within tile
  const int c0   = (lane < 16) ? 0 : 8;  // K interleave base per lane half
  const int hi   = lane >> 4;

  const int rowBase = blockIdx.x * 128 + wave * 16;
  const float* __restrict__ xrow = x + (size_t)(rowBase + m) * DMODEL;

  const v8f vzero = {0.f,0.f,0.f,0.f,0.f,0.f,0.f,0.f};
  v8f acc[4];
  #pragma unroll
  for (int nt = 0; nt < 4; ++nt) acc[nt] = vzero;

  for (int kk = 0; kk < DMODEL; kk += KCHUNK) {
    __syncthreads();
    // stage W[kk:kk+256, 0:64] transposed into LDS as f16
    #pragma unroll 4
    for (int idx = tid; idx < HEAD * KCHUNK; idx += 256) {
      const int kloc = idx >> 6;      // 0..255
      const int n    = idx & 63;
      lds_wt[n * KCHUNK + kloc] = (_Float16)W[(size_t)(kk + kloc) * HEAD + n];
    }
    __syncthreads();

    #pragma unroll
    for (int kc = 0; kc < KCHUNK; kc += 32) {
      const int k = kk + kc;
      // A fragment: 16 halves per lane (K = c0..c0+7, c0+16..c0+23)
      const float4 f0 = *(const float4*)(xrow + k + c0);
      const float4 f1 = *(const float4*)(xrow + k + c0 + 4);
      const float4 f2 = *(const float4*)(xrow + k + c0 + 16);
      const float4 f3 = *(const float4*)(xrow + k + c0 + 20);
      v16h a;
      a[0]=(_Float16)f0.x;  a[1]=(_Float16)f0.y;  a[2]=(_Float16)f0.z;  a[3]=(_Float16)f0.w;
      a[4]=(_Float16)f1.x;  a[5]=(_Float16)f1.y;  a[6]=(_Float16)f1.z;  a[7]=(_Float16)f1.w;
      a[8]=(_Float16)f2.x;  a[9]=(_Float16)f2.y;  a[10]=(_Float16)f2.z; a[11]=(_Float16)f2.w;
      a[12]=(_Float16)f3.x; a[13]=(_Float16)f3.y; a[14]=(_Float16)f3.z; a[15]=(_Float16)f3.w;

      // load ALL four B fragments first (8 ds_load_b128 clause), then WMMA
      v16h bfrag[4];
      #pragma unroll
      for (int nt = 0; nt < 4; ++nt) {
        const _Float16* wt = &lds_wt[(nt * 16 + m) * KCHUNK + kc];
        bfrag[nt] = cat16(*(const v8h*)(wt + c0), *(const v8h*)(wt + c0 + 16));
      }
      #pragma unroll
      for (int nt = 0; nt < 4; ++nt)
        acc[nt] = wmma_f16(a, bfrag[nt], acc[nt]);
    }
  }

  // C layout: lane (n = m), VGPR i -> row (i + 8*hi)
  if (which == 2) {
    #pragma unroll
    for (int nt = 0; nt < 4; ++nt)
      #pragma unroll
      for (int i = 0; i < 8; ++i) {
        const int row = rowBase + i + 8 * hi;
        const int b = row >> 12, s = row & (SEQ - 1);
        Vt[((size_t)(b * HEAD) + nt * 16 + m) * SEQ + s] = (_Float16)acc[nt][i];
      }
  } else {
    _Float16* __restrict__ out = (which == 0) ? Kf : Qf;
    #pragma unroll
    for (int nt = 0; nt < 4; ++nt)
      #pragma unroll
      for (int i = 0; i < 8; ++i) {
        const int row = rowBase + i + 8 * hi;
        out[(size_t)row * HEAD + nt * 16 + m] = (_Float16)acc[nt][i];
      }
  }
}

// ---------------------------------------------------------------------------
// Kernel 2: flash attention.  128 threads = 4 waves; each wave owns one
// 16-query tile of one batch.  Per 32-key step: 4 WMMAs produce two 16x16
// S tiles, ONE online-softmax update covers all 32 keys (half the shuffles /
// rescales of a per-16 update), V fragments are prefetched before softmax so
// their global latency hides under the VALU, then P goes through a per-wave
// LDS tile (C-layout -> A-layout) and one WMMA per output column tile.
// ---------------------------------------------------------------------------
__global__ __launch_bounds__(128) void attn_kernel(
    const _Float16* __restrict__ Qf, const _Float16* __restrict__ Kf,
    const _Float16* __restrict__ Vt, float* __restrict__ out)
{
  __shared__ _Float16 lds_p[4][16 * 32];   // per-wave P tile (16 q x 32 k)

  const int tid  = threadIdx.x;
  const int lane = tid & 31;
  const int wave = tid >> 5;
  const int m    = lane & 15;
  const int c0   = (lane < 16) ? 0 : 8;
  const int hi   = lane >> 4;

  const int b     = blockIdx.y;
  const int qtile = blockIdx.x * 4 + wave;

  // Q fragments (held for the whole kernel), pre-scaled by 1/sqrt(64) = 1/8
  const _Float16* __restrict__ qrow =
      Qf + (size_t)(b * SEQ + qtile * 16 + m) * HEAD;
  v16h aq0 = cat16(*(const v8h*)(qrow + c0),      *(const v8h*)(qrow + c0 + 16));
  v16h aq1 = cat16(*(const v8h*)(qrow + 32 + c0), *(const v8h*)(qrow + 32 + c0 + 16));
  #pragma unroll
  for (int j = 0; j < 16; ++j) {
    aq0[j] = aq0[j] * (_Float16)0.125f;
    aq1[j] = aq1[j] * (_Float16)0.125f;
  }

  const v8f vzero = {0.f,0.f,0.f,0.f,0.f,0.f,0.f,0.f};
  float mrun[8], lrun[8];
  v8f o[4];
  #pragma unroll
  for (int i = 0; i < 8; ++i) { mrun[i] = -1e30f; lrun[i] = 0.f; }
  #pragma unroll
  for (int nt = 0; nt < 4; ++nt) o[nt] = vzero;

  _Float16* __restrict__ ptile = &lds_p[wave][0];

  for (int kt = 0; kt < SEQ; kt += 32) {
    // ---- load all K fragments for both 16-key subtiles up front ----
    v16h bk[4];
    #pragma unroll
    for (int sub = 0; sub < 2; ++sub) {
      const _Float16* __restrict__ krow =
          Kf + (size_t)(b * SEQ + kt + sub * 16 + m) * HEAD;
      bk[2*sub]   = cat16(*(const v8h*)(krow + c0),      *(const v8h*)(krow + c0 + 16));
      bk[2*sub+1] = cat16(*(const v8h*)(krow + 32 + c0), *(const v8h*)(krow + 32 + c0 + 16));
    }
    v8f s0 = wmma_f16(aq1, bk[1], wmma_f16(aq0, bk[0], vzero));
    v8f s1 = wmma_f16(aq1, bk[3], wmma_f16(aq0, bk[2], vzero));

    // ---- prefetch V fragments (independent of softmax) ----
    v16h bv[4];
    #pragma unroll
    for (int nt = 0; nt < 4; ++nt) {
      const _Float16* __restrict__ vrow =
          Vt + (size_t)(b * HEAD + nt * 16 + m) * SEQ + kt;
      bv[nt] = cat16(*(const v8h*)(vrow + c0), *(const v8h*)(vrow + c0 + 16));
    }

    // ---- single online-softmax update across all 32 keys ----
    #pragma unroll
    for (int i = 0; i < 8; ++i) {
      const float v0 = s0[i], v1 = s1[i];
      float rmax = fmaxf(v0, v1);
      rmax = fmaxf(rmax, __shfl_xor(rmax, 1, 32));
      rmax = fmaxf(rmax, __shfl_xor(rmax, 2, 32));
      rmax = fmaxf(rmax, __shfl_xor(rmax, 4, 32));
      rmax = fmaxf(rmax, __shfl_xor(rmax, 8, 32));
      const float newm  = fmaxf(mrun[i], rmax);
      const float alpha = __expf(mrun[i] - newm);
      const float p0    = __expf(v0 - newm);
      const float p1    = __expf(v1 - newm);
      float psum = p0 + p1;
      psum += __shfl_xor(psum, 1, 32);
      psum += __shfl_xor(psum, 2, 32);
      psum += __shfl_xor(psum, 4, 32);
      psum += __shfl_xor(psum, 8, 32);
      mrun[i] = newm;
      lrun[i] = lrun[i] * alpha + psum;
      #pragma unroll
      for (int nt = 0; nt < 4; ++nt) o[nt][i] *= alpha;
      // C-layout -> LDS (row-major 16x32 f16 tile)
      ptile[(i + 8 * hi) * 32 + m]      = (_Float16)p0;
      ptile[(i + 8 * hi) * 32 + 16 + m] = (_Float16)p1;
    }

    // ---- reload P as an A fragment (LDS ops are in-order within a wave) ----
    const _Float16* prow = ptile + m * 32;
    v16h ap = cat16(*(const v8h*)(prow + c0), *(const v8h*)(prow + c0 + 16));

    // ---- O += P * V ----
    #pragma unroll
    for (int nt = 0; nt < 4; ++nt)
      o[nt] = wmma_f16(ap, bv[nt], o[nt]);
  }

  // normalize and write f32 output
  #pragma unroll
  for (int i = 0; i < 8; ++i) {
    const float rcp = 1.0f / lrun[i];
    const int row = qtile * 16 + i + 8 * hi;
    #pragma unroll
    for (int nt = 0; nt < 4; ++nt)
      out[(size_t)(b * SEQ + row) * HEAD + nt * 16 + m] = o[nt][i] * rcp;
  }
}

extern "C" void kernel_launch(void* const* d_in, const int* in_sizes, int n_in,
                              void* d_out, int out_size, void* d_ws, size_t ws_size,
                              hipStream_t stream) {
  (void)in_sizes; (void)n_in; (void)out_size; (void)ws_size;
  const float* x  = (const float*)d_in[0];
  const float* Wk = (const float*)d_in[1];
  const float* Wq = (const float*)d_in[2];
  const float* Wv = (const float*)d_in[3];

  // workspace: K (f16, row-major), Q (f16, row-major), V^T (f16, [b][64][4096])
  _Float16* Kf = (_Float16*)d_ws;
  _Float16* Qf = Kf + (size_t)NROWS * HEAD;
  _Float16* Vt = Qf + (size_t)NROWS * HEAD;

  dim3 g1(NROWS / 128, 3), b1(256);
  qkv_proj_kernel<<<g1, b1, 0, stream>>>(x, Wk, Wq, Wv, Kf, Qf, Vt);

  dim3 g2(SEQ / 16 / 4, 4), b2(128);
  attn_kernel<<<g2, b2, 0, stream>>>(Qf, Kf, Vt, (float*)d_out);
}
